// EGNN_NCP_46901042872369
// MI455X (gfx1250) — matrix operand order repaired and checked by reading, not compiled, not run on hardware
//
#include <hip/hip_runtime.h>
#include <hip/hip_bf16.h>

// ---------------------------------------------------------------------------
// EGNN forward for MI455X (gfx1250).  Edge MLP fused + bf16 WMMA +
// async global->LDS gather (ASYNCcnt path).
// ---------------------------------------------------------------------------

typedef __attribute__((ext_vector_type(16))) __bf16 bf16x16;
typedef __attribute__((ext_vector_type(8)))  float  f32x8;

#define KD   32
#define MD   32
#define NKER 5
#define MLPD 256
#define EIN_PAD   192   // 161 -> 6 * 32
#define HID_NPAD  336   // 322 -> 21 * 16  (GEMM1 N)
#define HID_KPAD  352   // 322 -> 11 * 32  (GEMM2 K)
#define EA_PAD    128   // 97  -> 128 (edge_attr pad)
#define ME        32    // edges per block in the edge-MLP kernel

__device__ __forceinline__ float silu_f(float x) {
    return x / (1.0f + __expf(-x));
}

union FragAB { uint4 q[2]; bf16x16 v; };

// A-fragment (16x32 bf16) from LDS, row-major with given stride.
// Per ISA layout, each lane's 16 elements are two contiguous 8-elem runs:
//   K = kBase + (lane>>4)*8 + {0..7}  and  + {16..23}
// -> exactly two ds_load_b128 per lane.
__device__ __forceinline__ bf16x16 load_a_frag(const __bf16* lds, int rowBase,
                                               int stride, int kBase, int lane) {
    const __bf16* p = lds + (rowBase + (lane & 15)) * stride + kBase + (lane >> 4) * 8;
    FragAB u;
    u.q[0] = *(const uint4*)p;          // elements 0..7
    u.q[1] = *(const uint4*)(p + 16);   // elements 8..15
    return u.v;
}

// B-fragment (32x16 bf16) from global weights stored [n][K] (K contiguous).
// lane&15 = column; K = kBase + (lane>>4)*8 + {0..7} and + {16..23}.
__device__ __forceinline__ bf16x16 load_b_frag(const __bf16* w, int nBase,
                                               int kStride, int kBase, int lane) {
    const __bf16* p = w + (size_t)(nBase + (lane & 15)) * kStride
                        + kBase + (lane >> 4) * 8;
    FragAB u;
    u.q[0] = *(const uint4*)p;          // 16B aligned: kStride % 8 == 0, kBase % 8 == 0
    u.q[1] = *(const uint4*)(p + 16);
    return u.v;
}

__device__ __forceinline__ f32x8 wmma_bf16(bf16x16 a, bf16x16 b, f32x8 c) {
    return __builtin_amdgcn_wmma_f32_16x16x32_bf16(false, a, false, b,
                                                   (short)0, c, false, false);
}

// Async global->LDS copy of 16 bytes per lane (CDNA5 ASYNCcnt path).
// vdst = LDS byte offset (wave-relative), vaddr = 64-bit global address.
__device__ __forceinline__ void async_g2l_b128(unsigned ldsOff, const void* g) {
    asm volatile("global_load_async_to_lds_b128 %0, %1, off"
                 :: "v"(ldsOff), "v"(g) : "memory");
}
__device__ __forceinline__ void wait_async0() {
    asm volatile("s_wait_asynccnt 0" ::: "memory");
}

__device__ __forceinline__ void layer_norm32(float* v, const float* g, const float* b) {
    float mu = 0.f;
#pragma unroll
    for (int i = 0; i < 32; ++i) mu += v[i];
    mu *= (1.0f / 32.0f);
    float var = 0.f;
#pragma unroll
    for (int i = 0; i < 32; ++i) { float d = v[i] - mu; var += d * d; }
    var *= (1.0f / 32.0f);
    float inv = rsqrtf(var + 1e-5f);
#pragma unroll
    for (int i = 0; i < 32; ++i) v[i] = (v[i] - mu) * inv * g[i] + b[i];
}

// ---------------------------------------------------------------------------
// K0: node embed  feats = props @ embed_W + b    (thread per (node, col))
// ---------------------------------------------------------------------------
__global__ void node_embed_kernel(const float* __restrict__ props,
                                  const float* __restrict__ W,
                                  const float* __restrict__ b,
                                  float* __restrict__ feats,
                                  __bf16* __restrict__ featsB,
                                  float* __restrict__ allf, int N_) {
    int idx = blockIdx.x * blockDim.x + threadIdx.x;
    int n = idx >> 5, c = idx & 31;
    if (n >= N_) return;
    float acc = b[c];
#pragma unroll
    for (int p = 0; p < 9; ++p) acc += props[n * 9 + p] * W[p * KD + c];
    feats[n * KD + c]  = acc;
    featsB[n * KD + c] = (__bf16)acc;
    allf[(size_t)n * (KD * (NKER + 1)) + c] = acc;
}

// ---------------------------------------------------------------------------
// K1: edge_attr = [sin(d_i), cos(d_i), |r|^2, edge_embed] padded to 128, bf16
// ---------------------------------------------------------------------------
__global__ void edge_attr_kernel(const float* __restrict__ pos,
                                 const float* __restrict__ eprops,
                                 const float* __restrict__ eW,
                                 const float* __restrict__ eb,
                                 const int* __restrict__ src,
                                 const int* __restrict__ dst,
                                 __bf16* __restrict__ eattr, int E_) {
    int e = blockIdx.x * blockDim.x + threadIdx.x;
    if (e >= E_) return;
    int s = src[e], d = dst[e];
    float dx = pos[s * 3 + 0] - pos[d * 3 + 0];
    float dy = pos[s * 3 + 1] - pos[d * 3 + 1];
    float dz = pos[s * 3 + 2] - pos[d * 3 + 2];
    float dist = dx * dx + dy * dy + dz * dz;
    __bf16* out = eattr + (size_t)e * EA_PAD;
#pragma unroll
    for (int i = 0; i < 32; ++i) {
        float di = dist * exp2f(-(float)i);
        out[i]      = (__bf16)__sinf(di);
        out[32 + i] = (__bf16)__cosf(di);
    }
    out[64] = (__bf16)dist;
#pragma unroll
    for (int c = 0; c < KD; ++c) {
        float acc = eb[c];
#pragma unroll
        for (int p = 0; p < 9; ++p) acc += eprops[(size_t)e * 9 + p] * eW[p * KD + c];
        out[65 + c] = (__bf16)acc;
    }
    for (int c = 97; c < EA_PAD; ++c) out[c] = (__bf16)0.f;
}

// ---------------------------------------------------------------------------
// K2: weight prep -> bf16, zero-padded, [n][K] layout (B-fragment friendly)
// ---------------------------------------------------------------------------
__global__ void prep_w1_kernel(const float* __restrict__ eW1, __bf16* __restrict__ W1b) {
    int idx = blockIdx.x * blockDim.x + threadIdx.x;
    const int per = HID_NPAD * EIN_PAD;
    if (idx >= NKER * per) return;
    int k = idx / per, r = idx % per, n = r / EIN_PAD, K = r % EIN_PAD;
    float v = (n < 322 && K < 161) ? eW1[(size_t)k * 161 * 322 + (size_t)K * 322 + n] : 0.f;
    W1b[idx] = (__bf16)v;
}

__global__ void prep_w2_kernel(const float* __restrict__ eW2, __bf16* __restrict__ W2b) {
    int idx = blockIdx.x * blockDim.x + threadIdx.x;
    const int per = MD * HID_KPAD;
    if (idx >= NKER * per) return;
    int k = idx / per, r = idx % per, n = r / HID_KPAD, K = r % HID_KPAD;
    float v = (K < 322) ? eW2[(size_t)k * 322 * MD + (size_t)K * MD + n] : 0.f;
    W2b[idx] = (__bf16)v;
}

// ---------------------------------------------------------------------------
// K3: degree
// ---------------------------------------------------------------------------
__global__ void deg_kernel(const int* __restrict__ dst, float* __restrict__ deg, int E_) {
    int e = blockIdx.x * blockDim.x + threadIdx.x;
    if (e < E_) atomicAdd(&deg[dst[e]], 1.0f);
}

// ---------------------------------------------------------------------------
// K4: fused edge MLP (WMMA bf16) + layernorm + scatter-mean numerator
//     block = 128 threads (4 waves), tile = 32 edges
// ---------------------------------------------------------------------------
__global__ __launch_bounds__(128)
void edge_mlp_kernel(const __bf16* __restrict__ featsB,
                     const __bf16* __restrict__ eattr,
                     const __bf16* __restrict__ W1b,
                     const __bf16* __restrict__ W2b,
                     const float* __restrict__ eb1,
                     const float* __restrict__ eb2,
                     const float* __restrict__ eln_g,
                     const float* __restrict__ eln_b,
                     const int* __restrict__ srcI,
                     const int* __restrict__ dstI,
                     float* __restrict__ miAccum, int kk, int E_) {
    __shared__ __align__(16) __bf16 lA[ME][EIN_PAD];   // 12 KB
    __shared__ __align__(16) __bf16 lH[ME][HID_KPAD];  // 22 KB
    __shared__ __align__(16) float  lM[ME][MD];        //  4 KB
    __shared__ int lDst[ME];

    const int tid  = threadIdx.x;
    const int lane = tid & 31;
    const int wave = tid >> 5;
    const int e0   = blockIdx.x * ME;

    for (int i = tid; i < ME * HID_KPAD; i += 128) ((__bf16*)lH)[i] = (__bf16)0.f;

    // Zero pad rows of lA only when this block has a tail (rare).
    if (e0 + ME > E_) {
        for (int i = tid; i < ME * EIN_PAD; i += 128) {
            int r = i / EIN_PAD;
            if (e0 + r >= E_) ((__bf16*)lA)[i] = (__bf16)0.f;
        }
    }

    // Async gather: [feats[dst] | feats[src] | edge_attr] -> [32,192] bf16.
    // 24 x 16B chunks per row; per-lane global addr + per-lane LDS offset.
    for (int i = tid; i < ME * 24; i += 128) {
        int r = i / 24, q = i % 24;
        int e = e0 + r;
        if (e < E_) {
            const char* g;
            if (q < 4)      g = (const char*)(featsB + (size_t)dstI[e] * KD) + q * 16;
            else if (q < 8) g = (const char*)(featsB + (size_t)srcI[e] * KD) + (q - 4) * 16;
            else            g = (const char*)(eattr + (size_t)e * EA_PAD) + (q - 8) * 16;
            unsigned ldsOff = (unsigned)(unsigned long long)(uintptr_t)&lA[r][q * 8];
            async_g2l_b128(ldsOff, g);
        }
    }
    if (tid < ME) { int e = e0 + tid; lDst[tid] = (e < E_) ? dstI[e] : 0; }
    wait_async0();
    __syncthreads();

    // GEMM1: [32,192] x [192,336] -> silu -> bf16 hidden in LDS
    const __bf16* W1 = W1b + (size_t)kk * HID_NPAD * EIN_PAD;
    for (int t = wave; t < 2 * (HID_NPAD / 16); t += 4) {
        int rt = t & 1, nt = t >> 1;
        f32x8 acc = {0.f, 0.f, 0.f, 0.f, 0.f, 0.f, 0.f, 0.f};
#pragma unroll
        for (int kc = 0; kc < EIN_PAD / 32; ++kc) {
            bf16x16 a = load_a_frag(&lA[0][0], rt * 16, EIN_PAD, kc * 32, lane);
            bf16x16 b = load_b_frag(W1, nt * 16, EIN_PAD, kc * 32, lane);
            acc = wmma_bf16(a, b, acc);
        }
        int col = nt * 16 + (lane & 15);
        float bias = (col < 322) ? eb1[kk * 322 + col] : 0.f;
#pragma unroll
        for (int i = 0; i < 8; ++i) {
            int row = rt * 16 + i + ((lane >= 16) ? 8 : 0);
            lH[row][col] = (__bf16)silu_f(acc[i] + bias);
        }
    }
    __syncthreads();

    // GEMM2: [32,352] x [352,32] -> silu -> lM
    const __bf16* W2 = W2b + (size_t)kk * MD * HID_KPAD;
    {
        int rt = wave & 1, nt = wave >> 1;   // 4 tiles, 4 waves
        f32x8 acc = {0.f, 0.f, 0.f, 0.f, 0.f, 0.f, 0.f, 0.f};
#pragma unroll
        for (int kc = 0; kc < HID_KPAD / 32; ++kc) {
            bf16x16 a = load_a_frag(&lH[0][0], rt * 16, HID_KPAD, kc * 32, lane);
            bf16x16 b = load_b_frag(W2, nt * 16, HID_KPAD, kc * 32, lane);
            acc = wmma_bf16(a, b, acc);
        }
        int col = nt * 16 + (lane & 15);
        float bias = eb2[kk * MD + col];
#pragma unroll
        for (int i = 0; i < 8; ++i) {
            int row = rt * 16 + i + ((lane >= 16) ? 8 : 0);
            lM[row][col] = silu_f(acc[i] + bias);
        }
    }
    __syncthreads();

    // Per-edge layernorm (wave32 = one row of 32) + atomic scatter to dst
    const float g = eln_g[kk * MD + lane];
    const float b = eln_b[kk * MD + lane];
    for (int r = wave; r < ME; r += 4) {
        float v  = lM[r][lane];
        float mu = v;
#pragma unroll
        for (int off = 16; off > 0; off >>= 1) mu += __shfl_xor(mu, off);
        mu *= (1.0f / 32.0f);
        float d   = v - mu;
        float var = d * d;
#pragma unroll
        for (int off = 16; off > 0; off >>= 1) var += __shfl_xor(var, off);
        var *= (1.0f / 32.0f);
        float y = d * rsqrtf(var + 1e-5f) * g + b;
        int e = e0 + r;
        if (e < E_) atomicAdd(&miAccum[(size_t)lDst[r] * MD + lane], y);
    }
}

// ---------------------------------------------------------------------------
// K5: node update (mean-agg LN, node MLP, residual).  Thread per node.
// ---------------------------------------------------------------------------
__global__ void node_update_kernel(float* __restrict__ feats,
                                   __bf16* __restrict__ featsB,
                                   float* __restrict__ allf,
                                   float* __restrict__ miAccum,
                                   const float* __restrict__ deg,
                                   const float* __restrict__ eln_g,
                                   const float* __restrict__ eln_b,
                                   const float* __restrict__ n1g,
                                   const float* __restrict__ n1b,
                                   const float* __restrict__ n2g,
                                   const float* __restrict__ n2b,
                                   const float* __restrict__ nW1,
                                   const float* __restrict__ nb1,
                                   const float* __restrict__ nW2,
                                   const float* __restrict__ nb2,
                                   int kk, int N_) {
    int n = blockIdx.x * blockDim.x + threadIdx.x;
    if (n >= N_) return;
    float x[64];            // [hf | m_i]
    float dg = fmaxf(deg[n], 1.0f);
#pragma unroll
    for (int c = 0; c < 32; ++c) {
        x[32 + c] = miAccum[(size_t)n * MD + c] / dg;
        miAccum[(size_t)n * MD + c] = 0.f;      // reset for next kernel iter
    }
    layer_norm32(&x[32], eln_g + kk * MD, eln_b + kk * MD);
#pragma unroll
    for (int c = 0; c < 32; ++c) x[c] = feats[(size_t)n * KD + c];
    layer_norm32(&x[0], n1g + kk * KD, n1b + kk * KD);

    float out[32];
#pragma unroll
    for (int c = 0; c < 32; ++c) out[c] = nb2[kk * KD + c];
    const float* W1 = nW1 + (size_t)kk * 64 * 64;
    const float* W2 = nW2 + (size_t)kk * 64 * 32;
    for (int j = 0; j < 64; ++j) {
        float h = nb1[kk * 64 + j];
#pragma unroll
        for (int i = 0; i < 64; ++i) h += x[i] * W1[i * 64 + j];
        h = silu_f(h);
#pragma unroll
        for (int c = 0; c < 32; ++c) out[c] += h * W2[j * 32 + c];
    }
    layer_norm32(out, n2g + kk * KD, n2b + kk * KD);
#pragma unroll
    for (int c = 0; c < 32; ++c) {
        float f = feats[(size_t)n * KD + c] + out[c];
        feats[(size_t)n * KD + c]  = f;
        featsB[(size_t)n * KD + c] = (__bf16)f;
        allf[(size_t)n * (KD * (NKER + 1)) + (kk + 1) * KD + c] = f;
    }
}

// ---------------------------------------------------------------------------
// K6: graph mean pool (numerator + counts via atomics)
// ---------------------------------------------------------------------------
__global__ void pool_kernel(const float* __restrict__ allf,
                            const int* __restrict__ batch,
                            float* __restrict__ gsum,
                            float* __restrict__ gcnt, int N_) {
    const int F = KD * (NKER + 1);
    int idx = blockIdx.x * blockDim.x + threadIdx.x;
    int n = idx / F, c = idx % F;
    if (n >= N_) return;
    int b = batch[n];
    atomicAdd(&gsum[(size_t)b * F + c], allf[(size_t)n * F + c]);
    if (c == 0) atomicAdd(&gcnt[b], 1.0f);
}

// ---------------------------------------------------------------------------
// K7: final FNN, one block per graph
// ---------------------------------------------------------------------------
__global__ __launch_bounds__(256)
void fnn_kernel(const float* __restrict__ gsum, const float* __restrict__ gcnt,
                const float* __restrict__ W0, const float* __restrict__ b0,
                const float* __restrict__ W1, const float* __restrict__ b1,
                const float* __restrict__ W2, const float* __restrict__ b2,
                const float* __restrict__ W3, const float* __restrict__ b3,
                float* __restrict__ out) {
    const int F = KD * (NKER + 1);
    __shared__ float g[F], h1[MLPD], h2[MLPD], h3[MLPD], red[8];
    int bidx = blockIdx.x, t = threadIdx.x;
    float cnt = fmaxf(gcnt[bidx], 1.0f);
    if (t < F) g[t] = gsum[(size_t)bidx * F + t] / cnt;
    __syncthreads();
    float a = b0[t];
    for (int i = 0; i < F; ++i) a += g[i] * W0[i * MLPD + t];
    h1[t] = silu_f(a);
    __syncthreads();
    a = b1[t];
    for (int i = 0; i < MLPD; ++i) a += h1[i] * W1[i * MLPD + t];
    h2[t] = silu_f(a);
    __syncthreads();
    a = b2[t];
    for (int i = 0; i < MLPD; ++i) a += h2[i] * W2[i * MLPD + t];
    h3[t] = silu_f(a);
    __syncthreads();
    float p = h3[t] * W3[t];
#pragma unroll
    for (int off = 16; off > 0; off >>= 1) p += __shfl_xor(p, off);
    if ((t & 31) == 0) red[t >> 5] = p;
    __syncthreads();
    if (t == 0) {
        float s = b3[0];
#pragma unroll
        for (int w = 0; w < 8; ++w) s += red[w];
        out[bidx] = s;
    }
}

// ---------------------------------------------------------------------------
// host launcher
// ---------------------------------------------------------------------------
static inline char* carve(char*& p, size_t bytes) {
    char* r = p;
    p += (bytes + 255) & ~size_t(255);
    return r;
}

extern "C" void kernel_launch(void* const* d_in, const int* in_sizes, int n_in,
                              void* d_out, int out_size, void* d_ws, size_t ws_size,
                              hipStream_t stream) {
    const float* props    = (const float*)d_in[0];
    const float* pos      = (const float*)d_in[1];
    const float* eprops   = (const float*)d_in[2];
    const float* embW     = (const float*)d_in[3];
    const float* embB     = (const float*)d_in[4];
    const float* eembW    = (const float*)d_in[5];
    const float* eembB    = (const float*)d_in[6];
    const float* k_eW1    = (const float*)d_in[7];
    const float* k_eb1    = (const float*)d_in[8];
    const float* k_eW2    = (const float*)d_in[9];
    const float* k_eb2    = (const float*)d_in[10];
    const float* k_eln_g  = (const float*)d_in[11];
    const float* k_eln_b  = (const float*)d_in[12];
    const float* k_n1g    = (const float*)d_in[13];
    const float* k_n1b    = (const float*)d_in[14];
    const float* k_n2g    = (const float*)d_in[15];
    const float* k_n2b    = (const float*)d_in[16];
    const float* k_nW1    = (const float*)d_in[17];
    const float* k_nb1    = (const float*)d_in[18];
    const float* k_nW2    = (const float*)d_in[19];
    const float* k_nb2    = (const float*)d_in[20];
    const float* fW0      = (const float*)d_in[21];
    const float* fb0      = (const float*)d_in[22];
    const float* fW1      = (const float*)d_in[23];
    const float* fb1      = (const float*)d_in[24];
    const float* fW2      = (const float*)d_in[25];
    const float* fb2      = (const float*)d_in[26];
    const float* fW3      = (const float*)d_in[27];
    const float* fb3      = (const float*)d_in[28];
    const int*   ei       = (const int*)d_in[29];
    const int*   batch    = (const int*)d_in[30];

    const int N_ = in_sizes[0] / 9;
    const int E_ = in_sizes[29] / 2;
    const int G_ = out_size;
    const int F  = KD * (NKER + 1);
    const int* srcI = ei;
    const int* dstI = ei + E_;

    char* p = (char*)d_ws;
    float*  feats   = (float*)carve(p, (size_t)N_ * KD * 4);
    float*  allf    = (float*)carve(p, (size_t)N_ * F * 4);
    float*  miAcc   = (float*)carve(p, (size_t)N_ * MD * 4);
    float*  deg     = (float*)carve(p, (size_t)N_ * 4);
    float*  gsum    = (float*)carve(p, (size_t)G_ * F * 4);
    float*  gcnt    = (float*)carve(p, (size_t)G_ * 4);
    __bf16* featsB  = (__bf16*)carve(p, (size_t)N_ * KD * 2);
    __bf16* eattr   = (__bf16*)carve(p, (size_t)E_ * EA_PAD * 2);
    __bf16* W1b     = (__bf16*)carve(p, (size_t)NKER * HID_NPAD * EIN_PAD * 2);
    __bf16* W2b     = (__bf16*)carve(p, (size_t)NKER * MD * HID_KPAD * 2);

    hipMemsetAsync(miAcc, 0, (size_t)N_ * MD * 4, stream);
    hipMemsetAsync(deg,   0, (size_t)N_ * 4, stream);
    hipMemsetAsync(gsum,  0, (size_t)G_ * F * 4, stream);
    hipMemsetAsync(gcnt,  0, (size_t)G_ * 4, stream);

    node_embed_kernel<<<(N_ * KD + 255) / 256, 256, 0, stream>>>(
        props, embW, embB, feats, featsB, allf, N_);
    edge_attr_kernel<<<(E_ + 255) / 256, 256, 0, stream>>>(
        pos, eprops, eembW, eembB, srcI, dstI, eattr, E_);
    prep_w1_kernel<<<(NKER * HID_NPAD * EIN_PAD + 255) / 256, 256, 0, stream>>>(k_eW1, W1b);
    prep_w2_kernel<<<(NKER * MD * HID_KPAD + 255) / 256, 256, 0, stream>>>(k_eW2, W2b);
    deg_kernel<<<(E_ + 255) / 256, 256, 0, stream>>>(dstI, deg, E_);

    for (int k = 0; k < NKER; ++k) {
        edge_mlp_kernel<<<(E_ + ME - 1) / ME, 128, 0, stream>>>(
            featsB, eattr, W1b, W2b, k_eb1, k_eb2, k_eln_g, k_eln_b,
            srcI, dstI, miAcc, k, E_);
        node_update_kernel<<<(N_ + 127) / 128, 128, 0, stream>>>(
            feats, featsB, allf, miAcc, deg, k_eln_g, k_eln_b,
            k_n1g, k_n1b, k_n2g, k_n2b, k_nW1, k_nb1, k_nW2, k_nb2, k, N_);
    }

    pool_kernel<<<((size_t)N_ * F + 255) / 256, 256, 0, stream>>>(allf, batch, gsum, gcnt, N_);
    fnn_kernel<<<G_, 256, 0, stream>>>(gsum, gcnt, fW0, fb0, fW1, fb1, fW2, fb2,
                                       fW3, fb3, (float*)d_out);
}
